// Block_FFD_Moe_42554535968925
// MI455X (gfx1250) — matrix operand order repaired and checked by reading, not compiled, not run on hardware
//
#include <hip/hip_runtime.h>
#include <stdint.h>

#define T_TOK 4096
#define DIN   1024
#define DOUT  1024
#define NEXP  8
#define HDIM  4096
#define ROWCAP (T_TOK * 2 + NEXP * 128) /* 9216: 8192 active rows + per-expert pad */
#define LDT 40                          /* LDS row stride in bf16 halves (32 + 8 pad) */

typedef __attribute__((ext_vector_type(16))) __bf16 v16bf;
typedef __attribute__((ext_vector_type(8)))  float  v8f;

union FragAB { uint4 q[2]; v16bf v; };

// Native f32 -> bf16 conversion (RNE): let the backend pick the bf16 cvt ops
// instead of a manual integer rounding sequence.
__device__ __forceinline__ unsigned short f2bf(float f) {
  union { __bf16 h; unsigned short u; } c;
  c.h = (__bf16)f;
  return c.u;
}
__device__ __forceinline__ unsigned pack2(float a, float b) {
  union { __bf16 h[2]; unsigned u; } c;
  c.h[0] = (__bf16)a;
  c.h[1] = (__bf16)b;
  return c.u;
}

// ---------------------------------------------------------------------------
// Kernel 1: noisy top-2 gating. One wave32 per token.
// ---------------------------------------------------------------------------
__global__ __launch_bounds__(256) void moe_gate(
    const float* __restrict__ x, const float* __restrict__ noise,
    const float* __restrict__ gate_w, const float* __restrict__ gate_b,
    const float* __restrict__ noise_w, const float* __restrict__ noise_b,
    float* __restrict__ wout) {
  const int lane = threadIdx.x & 31;
  const int wid  = threadIdx.x >> 5;
  const int t    = blockIdx.x * 8 + wid;

  float ag[8] = {0.f,0.f,0.f,0.f,0.f,0.f,0.f,0.f};
  float an[8] = {0.f,0.f,0.f,0.f,0.f,0.f,0.f,0.f};
  const float* xr = x + (size_t)t * DIN;
  for (int i = lane; i < DIN; i += 32) {
    float xv = xr[i];
    const float4* g4 = reinterpret_cast<const float4*>(gate_w + (size_t)i * NEXP);
    const float4* n4 = reinterpret_cast<const float4*>(noise_w + (size_t)i * NEXP);
    float4 g0 = g4[0], g1 = g4[1], m0 = n4[0], m1 = n4[1];
    ag[0] += xv * g0.x; ag[1] += xv * g0.y; ag[2] += xv * g0.z; ag[3] += xv * g0.w;
    ag[4] += xv * g1.x; ag[5] += xv * g1.y; ag[6] += xv * g1.z; ag[7] += xv * g1.w;
    an[0] += xv * m0.x; an[1] += xv * m0.y; an[2] += xv * m0.z; an[3] += xv * m0.w;
    an[4] += xv * m1.x; an[5] += xv * m1.y; an[6] += xv * m1.z; an[7] += xv * m1.w;
  }
#pragma unroll
  for (int e = 0; e < 8; ++e) {
    float a = ag[e], b = an[e];
#pragma unroll
    for (int off = 16; off > 0; off >>= 1) {
      a += __shfl_xor(a, off, 32);
      b += __shfl_xor(b, off, 32);
    }
    ag[e] = a; an[e] = b;
  }
  if (lane == 0) {
    float logit[8];
#pragma unroll
    for (int e = 0; e < 8; ++e) {
      float z  = an[e] + noise_b[e];
      float sp = (z > 20.f) ? z : log1pf(expf(z));   // softplus
      logit[e] = ag[e] + gate_b[e] + noise[(size_t)t * NEXP + e] * sp;
    }
    int i0 = 0; float v0 = logit[0];
#pragma unroll
    for (int e = 1; e < 8; ++e) if (logit[e] > v0) { v0 = logit[e]; i0 = e; }
    int i1 = (i0 == 0) ? 1 : 0; float v1 = logit[i1];
#pragma unroll
    for (int e = 0; e < 8; ++e)
      if (e != i0 && e > i1 && logit[e] > v1) { v1 = logit[e]; i1 = e; }
    float mx = fmaxf(v0, v1);
    float e0 = expf(v0 - mx), e1 = expf(v1 - mx);
    float inv = 1.f / (e0 + e1);
    float w[8] = {0.f,0.f,0.f,0.f,0.f,0.f,0.f,0.f};
    w[i0] = e0 * inv; w[i1] = e1 * inv;
#pragma unroll
    for (int e = 0; e < 8; ++e) wout[(size_t)t * NEXP + e] = w[e];
  }
}

// ---------------------------------------------------------------------------
// Kernel 2a: per-expert row counts (block e exclusively owns counts[e]).
// ---------------------------------------------------------------------------
__global__ __launch_bounds__(256) void moe_count(const float* __restrict__ wts,
                                                 int* __restrict__ counts) {
  __shared__ int red[256];
  const int e = blockIdx.x;
  int c = 0;
  for (int t = threadIdx.x; t < T_TOK; t += 256) c += (wts[(size_t)t * NEXP + e] != 0.f);
  red[threadIdx.x] = c; __syncthreads();
  for (int s = 128; s > 0; s >>= 1) {
    if (threadIdx.x < s) red[threadIdx.x] += red[threadIdx.x + s];
    __syncthreads();
  }
  if (threadIdx.x == 0) counts[e] = red[0];
}

// ---------------------------------------------------------------------------
// Kernel 2b: deterministic compaction into 128-aligned expert segments.
// ---------------------------------------------------------------------------
__global__ __launch_bounds__(256) void moe_build(const float* __restrict__ wts,
                                                 const int* __restrict__ counts,
                                                 int* __restrict__ row_token,
                                                 float* __restrict__ row_weight) {
  const int e = blockIdx.x;
  const int lane = threadIdx.x & 31, wid = threadIdx.x >> 5;
  __shared__ int wsum[8];
  __shared__ int woff[8];
  __shared__ int cbase;
  if (threadIdx.x == 0) cbase = 0;
  int po = 0;
  for (int i = 0; i < e; ++i) po += (counts[i] + 127) & ~127;
  __syncthreads();
  for (int chunk = 0; chunk < T_TOK / 256; ++chunk) {
    const int t = chunk * 256 + threadIdx.x;
    const float w = wts[(size_t)t * NEXP + e];
    const bool sel = (w != 0.f);
    const unsigned long long bal = __ballot(sel);
    const int lp = __popcll(bal & ((1ull << lane) - 1ull));
    if (lane == 0) wsum[wid] = __popcll(bal);
    __syncthreads();
    if (threadIdx.x == 0) {
      int s = cbase;
      for (int i = 0; i < 8; ++i) { woff[i] = s; s += wsum[i]; }
      cbase = s;
    }
    __syncthreads();
    if (sel) {
      const int slot = po + woff[wid] + lp;
      row_token[slot]  = t;
      row_weight[slot] = w;
    }
    __syncthreads();
  }
  const int cnt = counts[e], pad = (cnt + 127) & ~127;
  for (int i = cnt + threadIdx.x; i < pad; i += 256) {
    row_token[po + i]  = 0;
    row_weight[po + i] = 0.f;
  }
}

// ---------------------------------------------------------------------------
// Kernel 3: GEMM1  h = relu(gather(x) @ w1[e] + b1[e])  -> bf16 h buffer.
// Workgroup: 256 threads (8 waves), tile 128x128, K-step 32 (bf16 WMMA).
// Waves arranged 2(M) x 4(N); each wave owns 64x32 = 4x2 WMMA tiles.
// ---------------------------------------------------------------------------
__global__ __launch_bounds__(256) void moe_gemm1(
    const float* __restrict__ x, const float* __restrict__ w1,
    const float* __restrict__ b1, const int* __restrict__ counts,
    const int* __restrict__ row_token, unsigned short* __restrict__ hbuf) {
  const int e  = blockIdx.z;
  const int mt = blockIdx.x;
  const int nt = blockIdx.y;
  int po = 0;
#pragma unroll
  for (int i = 0; i < NEXP; ++i) {
    const int pc = (counts[i] + 127) & ~127;
    if (i < e) po += pc;
  }
  const int padded = (counts[e] + 127) & ~127;
  if (mt * 128 >= padded) return;
  const int row_base = po + mt * 128;
  const int n0 = nt * 128;

  __shared__ unsigned short As[128 * LDT];
  __shared__ unsigned short Bs[128 * LDT];
  __shared__ int toks[128];

  const int tid = threadIdx.x, lane = tid & 31, wid = tid >> 5;
  const int wm = wid >> 2, wn = wid & 3;
  if (tid < 128) toks[tid] = row_token[row_base + tid];
  __syncthreads();

  const v8f vzero = {0.f,0.f,0.f,0.f,0.f,0.f,0.f,0.f};
  v8f acc[4][2];
#pragma unroll
  for (int mi = 0; mi < 4; ++mi)
#pragma unroll
    for (int ni = 0; ni < 2; ++ni) acc[mi][ni] = vzero;

  for (int k0 = 0; k0 < DIN; k0 += 32) {
    // A tile: 128 gathered token rows x 32 k, f32 -> bf16
#pragma unroll
    for (int i = 0; i < 4; ++i) {
      const int task = tid + i * 256;
      const int row = task >> 3;
      const int kk  = (task & 7) * 4;
      const float4 v = *reinterpret_cast<const float4*>(
          x + (size_t)toks[row] * DIN + k0 + kk);
      *reinterpret_cast<uint2*>(&As[row * LDT + kk]) =
          make_uint2(pack2(v.x, v.y), pack2(v.z, v.w));
    }
    // B tile: w1[e][k0:k0+32, n0:n0+128], stored LDS-transposed [n][k]
#pragma unroll
    for (int i = 0; i < 4; ++i) {
      const int task = tid + i * 256;
      const int k  = task >> 5;
      const int nn = (task & 31) * 4;
      const float4 v = *reinterpret_cast<const float4*>(
          w1 + (size_t)e * DIN * HDIM + (size_t)(k0 + k) * HDIM + n0 + nn);
      Bs[(nn + 0) * LDT + k] = f2bf(v.x);
      Bs[(nn + 1) * LDT + k] = f2bf(v.y);
      Bs[(nn + 2) * LDT + k] = f2bf(v.z);
      Bs[(nn + 3) * LDT + k] = f2bf(v.w);
    }
    // prefetch next K-tile of the weight stream (global_prefetch_b8)
    if (k0 + 32 < DIN) {
      __builtin_prefetch(w1 + (size_t)e * DIN * HDIM +
                             (size_t)(k0 + 32 + (tid >> 5)) * HDIM +
                             n0 + (tid & 31) * 4, 0, 1);
    }
    __syncthreads();

    FragAB a[4], b[2];
    const int klA = (lane >> 4) * 8;   // A: K 0..7 / 8..15 halves (+16 for hi)
    const int klB = (lane >> 4) * 8;   // B: K 0..7 / 8..15 halves (+16 for hi)
#pragma unroll
    for (int mi = 0; mi < 4; ++mi) {
      const int m = wm * 64 + mi * 16 + (lane & 15);
      a[mi].q[0] = *reinterpret_cast<const uint4*>(&As[m * LDT + klA]);
      a[mi].q[1] = *reinterpret_cast<const uint4*>(&As[m * LDT + klA + 16]);
    }
#pragma unroll
    for (int ni = 0; ni < 2; ++ni) {
      const int n = wn * 32 + ni * 16 + (lane & 15);
      b[ni].q[0] = *reinterpret_cast<const uint4*>(&Bs[n * LDT + klB]);
      b[ni].q[1] = *reinterpret_cast<const uint4*>(&Bs[n * LDT + klB + 16]);
    }
#pragma unroll
    for (int mi = 0; mi < 4; ++mi)
#pragma unroll
      for (int ni = 0; ni < 2; ++ni)
        acc[mi][ni] = __builtin_amdgcn_wmma_f32_16x16x32_bf16(
            false, a[mi].v, false, b[ni].v, (short)0, acc[mi][ni], false, false);
    __syncthreads();
  }

  // epilogue: +b1, relu, bf16 store to h (bias hoisted: depends on ni/lane only)
  float bia[2];
#pragma unroll
  for (int ni = 0; ni < 2; ++ni)
    bia[ni] = b1[(size_t)e * HDIM + n0 + wn * 32 + ni * 16 + (lane & 15)];
#pragma unroll
  for (int mi = 0; mi < 4; ++mi)
#pragma unroll
    for (int ni = 0; ni < 2; ++ni)
#pragma unroll
      for (int r = 0; r < 8; ++r) {
        const int m = wm * 64 + mi * 16 + r + ((lane >> 4) << 3);
        const int c = n0 + wn * 32 + ni * 16 + (lane & 15);
        float v = acc[mi][ni][r] + bia[ni];
        v = v > 0.f ? v : 0.f;
        hbuf[(size_t)(row_base + m) * HDIM + c] = f2bf(v);
      }
}

// ---------------------------------------------------------------------------
// Kernel 4: GEMM2  out = h @ w2[e] + b2[e], scaled by gate weight,
// atomically combined into x_out (exactly 2 adds per element -> deterministic).
// A tile (already bf16) is staged with GLOBAL_LOAD_ASYNC_TO_LDS_B64
// (ASYNCcnt-tracked DMA, no VGPR round-trip), fenced by s_wait_asynccnt.
// ---------------------------------------------------------------------------
__global__ __launch_bounds__(256) void moe_gemm2(
    const unsigned short* __restrict__ hbuf, const float* __restrict__ w2,
    const float* __restrict__ b2, const int* __restrict__ counts,
    const int* __restrict__ row_token, const float* __restrict__ row_weight,
    float* __restrict__ xout) {
  const int rt = blockIdx.x, nt = blockIdx.y;
  const int row0 = rt * 128;
  int total = 0, e = -1;
#pragma unroll
  for (int i = 0; i < NEXP; ++i) {
    const int pc = (counts[i] + 127) & ~127;
    if (row0 >= total && row0 < total + pc) e = i;
    total += pc;
  }
  if (row0 >= total) return;
  const int n0 = nt * 128;

  __shared__ unsigned short As[128 * LDT];
  __shared__ unsigned short Bs[128 * LDT];
  __shared__ int   toks[128];
  __shared__ float wgt[128];

  const int tid = threadIdx.x, lane = tid & 31, wid = tid >> 5;
  const int wm = wid >> 2, wn = wid & 3;
  if (tid < 128) {
    toks[tid] = row_token[row0 + tid];
    wgt[tid]  = row_weight[row0 + tid];
  }
  const unsigned long long hbase = (unsigned long long)(uintptr_t)hbuf;
  __syncthreads();

  const v8f vzero = {0.f,0.f,0.f,0.f,0.f,0.f,0.f,0.f};
  v8f acc[4][2];
#pragma unroll
  for (int mi = 0; mi < 4; ++mi)
#pragma unroll
    for (int ni = 0; ni < 2; ++ni) acc[mi][ni] = vzero;

  for (int k0 = 0; k0 < HDIM; k0 += 32) {
    // A tile: h rows are already bf16 -> async DMA straight into LDS.
    // GVS addressing: mem = SGPR64(hbuf) + VGPR32(byte offset); dsaddr = VGPR.
#pragma unroll
    for (int i = 0; i < 4; ++i) {
      const int task = tid + i * 256;
      const int row = task >> 3;
      const int kk  = (task & 7) * 4;
      const unsigned lds_addr =
          (unsigned)(uintptr_t)(void*)&As[row * LDT + kk];
      const unsigned goff =
          (unsigned)(((size_t)(row0 + row) * HDIM + (size_t)(k0 + kk)) * 2u);
      asm volatile("global_load_async_to_lds_b64 %0, %1, %2"
                   :
                   : "v"(lds_addr), "v"(goff), "s"(hbase)
                   : "memory");
    }
    // B tile: w2[e][k0:k0+32, n0:n0+128] f32 -> bf16, LDS [n][k]
#pragma unroll
    for (int i = 0; i < 4; ++i) {
      const int task = tid + i * 256;
      const int k  = task >> 5;
      const int nn = (task & 31) * 4;
      const float4 v = *reinterpret_cast<const float4*>(
          w2 + (size_t)e * HDIM * DOUT + (size_t)(k0 + k) * DOUT + n0 + nn);
      Bs[(nn + 0) * LDT + k] = f2bf(v.x);
      Bs[(nn + 1) * LDT + k] = f2bf(v.y);
      Bs[(nn + 2) * LDT + k] = f2bf(v.z);
      Bs[(nn + 3) * LDT + k] = f2bf(v.w);
    }
    // prefetch next K-tile of w2 (global_prefetch_b8)
    if (k0 + 32 < HDIM) {
      __builtin_prefetch(w2 + (size_t)e * HDIM * DOUT +
                             (size_t)(k0 + 32 + (tid >> 5)) * DOUT +
                             n0 + (tid & 31) * 4, 0, 1);
    }
    asm volatile("s_wait_asynccnt 0x0" ::: "memory");
    __syncthreads();

    FragAB a[4], b[2];
    const int klA = (lane >> 4) * 8;
    const int klB = (lane >> 4) * 8;
#pragma unroll
    for (int mi = 0; mi < 4; ++mi) {
      const int m = wm * 64 + mi * 16 + (lane & 15);
      a[mi].q[0] = *reinterpret_cast<const uint4*>(&As[m * LDT + klA]);
      a[mi].q[1] = *reinterpret_cast<const uint4*>(&As[m * LDT + klA + 16]);
    }
#pragma unroll
    for (int ni = 0; ni < 2; ++ni) {
      const int n = wn * 32 + ni * 16 + (lane & 15);
      b[ni].q[0] = *reinterpret_cast<const uint4*>(&Bs[n * LDT + klB]);
      b[ni].q[1] = *reinterpret_cast<const uint4*>(&Bs[n * LDT + klB + 16]);
    }
#pragma unroll
    for (int mi = 0; mi < 4; ++mi)
#pragma unroll
      for (int ni = 0; ni < 2; ++ni)
        acc[mi][ni] = __builtin_amdgcn_wmma_f32_16x16x32_bf16(
            false, a[mi].v, false, b[ni].v, (short)0, acc[mi][ni], false, false);
    __syncthreads();
  }

  // epilogue: +b2 (hoisted), scale by gate weight, combine
  float bia[2];
#pragma unroll
  for (int ni = 0; ni < 2; ++ni)
    bia[ni] = b2[(size_t)e * DOUT + n0 + wn * 32 + ni * 16 + (lane & 15)];
#pragma unroll
  for (int mi = 0; mi < 4; ++mi)
#pragma unroll
    for (int ni = 0; ni < 2; ++ni)
#pragma unroll
      for (int r = 0; r < 8; ++r) {
        const int m = wm * 64 + mi * 16 + r + ((lane >> 4) << 3);
        const int c = n0 + wn * 32 + ni * 16 + (lane & 15);
        const float w = wgt[m];
        if (w != 0.f) {
          const float v = acc[mi][ni][r] + bia[ni];
          atomicAdd(&xout[(size_t)toks[m] * DOUT + c], w * v);
        }
      }
}

// ---------------------------------------------------------------------------
extern "C" void kernel_launch(void* const* d_in, const int* in_sizes, int n_in,
                              void* d_out, int out_size, void* d_ws, size_t ws_size,
                              hipStream_t stream) {
  (void)in_sizes; (void)n_in; (void)out_size; (void)ws_size;
  const float* x       = (const float*)d_in[0];
  const float* noise   = (const float*)d_in[1];
  const float* gate_w  = (const float*)d_in[2];
  const float* gate_b  = (const float*)d_in[3];
  const float* noise_w = (const float*)d_in[4];
  const float* noise_b = (const float*)d_in[5];
  const float* w1      = (const float*)d_in[6];
  const float* b1      = (const float*)d_in[7];
  const float* w2      = (const float*)d_in[8];
  const float* b2      = (const float*)d_in[9];

  float* xout = (float*)d_out;                       // [T, DOUT]
  float* wout = xout + (size_t)T_TOK * DOUT;         // [T, E] gate weights output

  // workspace carve-up
  int*   counts     = (int*)d_ws;                    // [16]
  int*   row_token  = counts + 16;                   // [ROWCAP]
  float* row_weight = (float*)(row_token + ROWCAP);  // [ROWCAP]
  unsigned short* hbuf =
      (unsigned short*)((char*)d_ws + (128 << 10));  // [ROWCAP, HDIM] bf16 (~72MB, L2-resident)

  hipMemsetAsync(xout, 0, (size_t)T_TOK * DOUT * sizeof(float), stream);

  moe_gate<<<T_TOK / 8, 256, 0, stream>>>(x, noise, gate_w, gate_b, noise_w, noise_b, wout);
  moe_count<<<NEXP, 256, 0, stream>>>(wout, counts);
  moe_build<<<NEXP, 256, 0, stream>>>(wout, counts, row_token, row_weight);
  moe_gemm1<<<dim3(T_TOK / 128, HDIM / 128, NEXP), 256, 0, stream>>>(
      x, w1, b1, counts, row_token, hbuf);
  moe_gemm2<<<dim3(ROWCAP / 128, DOUT / 128), 256, 0, stream>>>(
      hbuf, w2, b2, counts, row_token, row_weight, xout);
}
